// Encoder_38946763440231
// MI455X (gfx1250) — matrix operand order repaired and checked by reading, not compiled, not run on hardware
//
#include <hip/hip_runtime.h>
#include <hip/hip_bf16.h>

#define N_NODES 100000
#define N_EDGES 1600000
#define IN_FEATS 256
#define NH 128
#define M_TILES (N_NODES / 16)   // 6250, exact
#define SCAN_B 256
#define BPAD 8                   // LDS row pad (bf16 elems) -> conflict-free ds_load_b128

typedef __attribute__((ext_vector_type(16))) __bf16 v16bf;
typedef __attribute__((ext_vector_type(8)))  float  v8f;

union Frag {
    v16bf v;
    struct { uint4 lo, hi; } q;
};

// ---------------- utility kernels ----------------

__global__ void k_zero(float* __restrict__ p, int n) {
    int i = blockIdx.x * blockDim.x + threadIdx.x;
    if (i < n) p[i] = 0.0f;
}

// accumulate float degrees (for norms) + int in-degree (for CSR)
__global__ void k_degree(const int* __restrict__ src, const int* __restrict__ dst,
                         float* __restrict__ degOut, float* __restrict__ degIn,
                         int* __restrict__ degI, int E) {
    int e = blockIdx.x * blockDim.x + threadIdx.x;
    if (e < E) {
        atomicAdd(&degOut[src[e]], 1.0f);
        atomicAdd(&degIn[dst[e]], 1.0f);
        atomicAdd(&degI[dst[e]], 1);
    }
}

// in-place deg -> D^{-1/2} with zero-degree -> 0
__global__ void k_norm(float* __restrict__ d, int n) {
    int i = blockIdx.x * blockDim.x + threadIdx.x;
    if (i < n) {
        float v = d[i];
        d[i] = (v > 0.0f) ? rsqrtf(v) : 0.0f;
    }
}

// ---- 3-stage exclusive scan of int degrees -> rowptr[N+1] ----
__global__ void k_scan1(const int* __restrict__ deg, int* __restrict__ part,
                        int* __restrict__ sums, int n) {
    __shared__ int tmp[SCAN_B];
    int i = blockIdx.x * SCAN_B + threadIdx.x;
    tmp[threadIdx.x] = (i < n) ? deg[i] : 0;
    __syncthreads();
    for (int d = 1; d < SCAN_B; d <<= 1) {
        int t = (threadIdx.x >= d) ? tmp[threadIdx.x - d] : 0;
        __syncthreads();
        tmp[threadIdx.x] += t;
        __syncthreads();
    }
    if (i < n) part[i] = tmp[threadIdx.x];                 // inclusive
    if (threadIdx.x == SCAN_B - 1) sums[blockIdx.x] = tmp[SCAN_B - 1];
}

__global__ void k_scan2(int* __restrict__ sums, int nb) {   // single block, nb <= 1024
    __shared__ int tmp[1024];
    int i = threadIdx.x;
    tmp[i] = (i < nb) ? sums[i] : 0;
    __syncthreads();
    for (int d = 1; d < 1024; d <<= 1) {
        int t = (i >= d) ? tmp[i - d] : 0;
        __syncthreads();
        tmp[i] += t;
        __syncthreads();
    }
    if (i < nb) sums[i] = tmp[i];                           // inclusive
}

__global__ void k_scan3(const int* __restrict__ part, const int* __restrict__ sums,
                        int* __restrict__ rowptr, int n) {
    int i = blockIdx.x * SCAN_B + threadIdx.x;
    if (i < n) {
        int incl = part[i] + (blockIdx.x > 0 ? sums[blockIdx.x - 1] : 0);
        rowptr[i + 1] = incl;
        if (i == 0) rowptr[0] = 0;
    }
}

__global__ void k_csr_fill(const int* __restrict__ src, const int* __restrict__ dst,
                           const int* __restrict__ rowptr, int* __restrict__ cursor,
                           int* __restrict__ col, int E) {
    int e = blockIdx.x * blockDim.x + threadIdx.x;
    if (e < E) {
        int d = dst[e];
        int p = atomicAdd(&cursor[d], 1);
        col[rowptr[d] + p] = src[e];
    }
}

// W [K x NH] f32 row-major  ->  Wt [NH x K] bf16 row-major
__global__ void k_wt(const float* __restrict__ W, __bf16* __restrict__ Wt, int K) {
    int idx = blockIdx.x * blockDim.x + threadIdx.x;
    if (idx < K * NH) {
        int n = idx / K;
        int k = idx % K;
        Wt[idx] = (__bf16)W[k * NH + n];
    }
}

// h = bf16(features[perm])  (layer-0 staging: fused permute + downconvert)
__global__ void k_permute_bf16(const float* __restrict__ feats, const int* __restrict__ perm,
                               __bf16* __restrict__ out) {
    int idx = blockIdx.x * blockDim.x + threadIdx.x;
    if (idx < N_NODES * IN_FEATS) {
        int i = idx >> 8;
        int k = idx & (IN_FEATS - 1);
        out[idx] = (__bf16)feats[(size_t)perm[i] * IN_FEATS + k];
    }
}

// ---------------- pull-mode aggregation, fused epilogue ----------------
// one wave per node; lane owns a float4 column slice; neighbor idx broadcast via shfl.
// out = [relu]((sum_{s in N(n)} hw[s,:]) * ns_in[n] + b)
__global__ void k_aggregate(const int* __restrict__ rowptr, const int* __restrict__ col,
                            const float* __restrict__ hw, const float* __restrict__ nin,
                            const float* __restrict__ bias, __bf16* __restrict__ outB,
                            float* __restrict__ outF, int relu) {
    int wid = (blockIdx.x * blockDim.x + threadIdx.x) >> 5;   // node id (wave-uniform)
    if (wid >= N_NODES) return;
    int lane = threadIdx.x & 31;
    int c = lane << 2;                                        // column chunk 0..124

    int s0 = rowptr[wid];
    int s1 = rowptr[wid + 1];

    float4 acc = make_float4(0.f, 0.f, 0.f, 0.f);
    for (int base = s0; base < s1; base += 32) {
        int cnt = min(32, s1 - base);
        int myIdx = (base + lane < s1) ? col[base + lane] : 0;
        // warm L2 for the rows this wave is about to gather
        __builtin_prefetch(hw + (size_t)myIdx * NH + c, 0, 3);
        for (int t = 0; t < cnt; ++t) {
            int s = __shfl(myIdx, t, 32);
            float4 v = *(const float4*)(hw + (size_t)s * NH + c);
            acc.x += v.x; acc.y += v.y; acc.z += v.z; acc.w += v.w;
        }
    }

    float nv = nin[wid];
    float4 bb = *(const float4*)(bias + c);
    float4 o;
    o.x = acc.x * nv + bb.x;
    o.y = acc.y * nv + bb.y;
    o.z = acc.z * nv + bb.z;
    o.w = acc.w * nv + bb.w;
    if (relu) {
        o.x = fmaxf(o.x, 0.f); o.y = fmaxf(o.y, 0.f);
        o.z = fmaxf(o.z, 0.f); o.w = fmaxf(o.w, 0.f);
    }
    if (outB) {
        union { __bf16 h[4]; unsigned long long u; } p;
        p.h[0] = (__bf16)o.x; p.h[1] = (__bf16)o.y;
        p.h[2] = (__bf16)o.z; p.h[3] = (__bf16)o.w;
        *(unsigned long long*)(outB + (size_t)wid * NH + c) = p.u;
    }
    if (outF) *(float4*)(outF + (size_t)wid * NH + c) = o;
}

// ---------------- WMMA GEMM with LDS-staged weights ----------------
// C[m, 0..127] = (A[m, :K] @ Wt^T) * nrm[m]
// A: [M x K] bf16 row-major; Wt: [NH x K] bf16 row-major (W transposed).
// Block = 4 waves; Wt staged in LDS once per block (row pad -> conflict-free b128).
extern __shared__ char smem[];
__global__ void k_gemm_wmma(const __bf16* __restrict__ A, const __bf16* __restrict__ Wt,
                            const float* __restrict__ nrm, float* __restrict__ C, int K) {
    __bf16* Bs = (__bf16*)smem;
    const int KP = K + BPAD;

    // cooperative stage of Wt -> LDS (uint4 = 8 bf16 per copy)
    const int chunksPerRow = K >> 3;
    for (int t = threadIdx.x; t < NH * chunksPerRow; t += blockDim.x) {
        int n = t / chunksPerRow;
        int kc = t % chunksPerRow;
        *(uint4*)(Bs + n * KP + kc * 8) = *(const uint4*)(Wt + n * K + kc * 8);
    }
    __syncthreads();

    const int lane = threadIdx.x & 31;
    const int wave = threadIdx.x >> 5;
    const int mt   = blockIdx.x * 4 + wave;
    if (mt >= M_TILES) return;            // wave-uniform: EXEC all-1s at WMMA

    const int half = lane >> 4;
    const int lm   = lane & 15;
    const int m0   = mt * 16;

    v8f zero = {};
    v8f acc[8];
#pragma unroll
    for (int i = 0; i < 8; ++i) acc[i] = zero;

    const __bf16* arow = A + (size_t)(m0 + lm) * K;

    // A fragment double-buffer: prefetch next K-step while WMMAs consume current
    Frag a, an;
    a.q.lo = *(const uint4*)(arow + 8 * half);
    a.q.hi = *(const uint4*)(arow + 16 + 8 * half);
    for (int k0 = 0; k0 < K; k0 += 32) {
        if (k0 + 32 < K) {
            an.q.lo = *(const uint4*)(arow + k0 + 32 + 8 * half);
            an.q.hi = *(const uint4*)(arow + k0 + 48 + 8 * half);
        }
#pragma unroll
        for (int nt = 0; nt < 8; ++nt) {
            const __bf16* bp = Bs + (nt * 16 + lm) * KP + k0 + 8 * half;
            Frag b;
            b.q.lo = *(const uint4*)(bp);
            b.q.hi = *(const uint4*)(bp + 16);
            acc[nt] = __builtin_amdgcn_wmma_f32_16x16x32_bf16(
                false, a.v, false, b.v, (short)0, acc[nt], false, false);
        }
        a = an;
    }

    // C/D 16x16 f32 layout: VGPR r -> row (r + 8*half), col = lane%16; fuse D_out^{-1/2}
#pragma unroll
    for (int nt = 0; nt < 8; ++nt) {
#pragma unroll
        for (int r = 0; r < 8; ++r) {
            int m = m0 + r + 8 * half;
            C[(size_t)m * NH + nt * 16 + lm] = acc[nt][r] * nrm[m];
        }
    }
}

// ---------------- host orchestration ----------------

extern "C" void kernel_launch(void* const* d_in, const int* in_sizes, int n_in,
                              void* d_out, int out_size, void* d_ws, size_t ws_size,
                              hipStream_t stream) {
    const float* feats = (const float*)d_in[0];
    const int*   src   = (const int*)d_in[1];
    const int*   dst   = (const int*)d_in[2];
    const int*   perm  = (const int*)d_in[3];
    const float* W0    = (const float*)d_in[4];
    const float* b0    = (const float*)d_in[5];
    const float* W1    = (const float*)d_in[6];
    const float* b1    = (const float*)d_in[7];
    const float* W2    = (const float*)d_in[8];
    const float* b2    = (const float*)d_in[9];
    float* out = (float*)d_out;           // N_NODES x NH f32

    char* w = (char*)d_ws;
    size_t off = 0;
    auto carve = [&](size_t bytes) {
        void* p = w + off;
        off = (off + bytes + 255) & ~(size_t)255;
        return p;
    };
    float*  norm_out = (float*)carve(sizeof(float) * N_NODES);
    float*  norm_in  = (float*)carve(sizeof(float) * N_NODES);
    int*    degI     = (int*)carve(sizeof(int) * N_NODES);
    int*    part     = (int*)carve(sizeof(int) * N_NODES);
    int*    bsum     = (int*)carve(sizeof(int) * 1024);
    int*    rowptr   = (int*)carve(sizeof(int) * (N_NODES + 1));
    int*    cursor   = (int*)carve(sizeof(int) * N_NODES);
    int*    col      = (int*)carve(sizeof(int) * N_EDGES);
    __bf16* Wt0      = (__bf16*)carve(sizeof(__bf16) * IN_FEATS * NH);
    __bf16* Wt1      = (__bf16*)carve(sizeof(__bf16) * NH * NH);
    __bf16* Wt2      = (__bf16*)carve(sizeof(__bf16) * NH * NH);
    __bf16* hb       = (__bf16*)carve(sizeof(__bf16) * (size_t)N_NODES * IN_FEATS);
    float*  hw       = (float*)carve(sizeof(float) * (size_t)N_NODES * NH);
    (void)ws_size; (void)in_sizes; (void)n_in; (void)out_size;

    const int TB = 256;
    const int nodeBlk = (N_NODES + TB - 1) / TB;           // 391
    const int edgeBlk = (N_EDGES + TB - 1) / TB;
    const int elInBlk = (N_NODES * IN_FEATS + TB - 1) / TB;
    const int aggBlk  = (N_NODES * 32 + TB - 1) / TB;      // wave per node, 8 waves/block
    const int gemmBlk = (M_TILES + 3) / 4;                 // 4 waves/block, 1 M-tile/wave

    // degrees -> norms + int in-degree
    k_zero<<<nodeBlk, TB, 0, stream>>>(norm_out, N_NODES);
    k_zero<<<nodeBlk, TB, 0, stream>>>(norm_in, N_NODES);
    k_zero<<<nodeBlk, TB, 0, stream>>>((float*)degI, N_NODES);
    k_zero<<<nodeBlk, TB, 0, stream>>>((float*)cursor, N_NODES);
    k_degree<<<edgeBlk, TB, 0, stream>>>(src, dst, norm_out, norm_in, degI, N_EDGES);
    k_norm<<<nodeBlk, TB, 0, stream>>>(norm_out, N_NODES);
    k_norm<<<nodeBlk, TB, 0, stream>>>(norm_in, N_NODES);

    // CSR build (once, reused by all 3 layers)
    k_scan1<<<nodeBlk, SCAN_B, 0, stream>>>(degI, part, bsum, N_NODES);
    k_scan2<<<1, 1024, 0, stream>>>(bsum, nodeBlk);
    k_scan3<<<nodeBlk, SCAN_B, 0, stream>>>(part, bsum, rowptr, N_NODES);
    k_csr_fill<<<edgeBlk, TB, 0, stream>>>(src, dst, rowptr, cursor, col, N_EDGES);

    // weights -> bf16 transposed
    k_wt<<<(IN_FEATS * NH + TB - 1) / TB, TB, 0, stream>>>(W0, Wt0, IN_FEATS);
    k_wt<<<(NH * NH + TB - 1) / TB, TB, 0, stream>>>(W1, Wt1, NH);
    k_wt<<<(NH * NH + TB - 1) / TB, TB, 0, stream>>>(W2, Wt2, NH);

    // layer-0 input: corrupt (permute) + downconvert
    k_permute_bf16<<<elInBlk, TB, 0, stream>>>(feats, perm, hb);

    const size_t lds0 = sizeof(__bf16) * NH * (IN_FEATS + BPAD);
    const size_t lds1 = sizeof(__bf16) * NH * (NH + BPAD);

    // ---- layer 0 ----
    k_gemm_wmma<<<gemmBlk, 128, lds0, stream>>>(hb, Wt0, norm_out, hw, IN_FEATS);
    k_aggregate<<<aggBlk, TB, 0, stream>>>(rowptr, col, hw, norm_in, b0, hb, nullptr, 1);

    // ---- layer 1 ----
    k_gemm_wmma<<<gemmBlk, 128, lds1, stream>>>(hb, Wt1, norm_out, hw, NH);
    k_aggregate<<<aggBlk, TB, 0, stream>>>(rowptr, col, hw, norm_in, b1, hb, nullptr, 1);

    // ---- layer 2 (no relu, f32 final output) ----
    k_gemm_wmma<<<gemmBlk, 128, lds1, stream>>>(hb, Wt2, norm_out, hw, NH);
    k_aggregate<<<aggBlk, TB, 0, stream>>>(rowptr, col, hw, norm_in, b2, nullptr, out, 0);
}
